// EBM_33913061769498
// MI455X (gfx1250) — matrix operand order repaired and checked by reading, not compiled, not run on hardware
//
#include <hip/hip_runtime.h>
#include <stdint.h>

typedef _Float16 v16h __attribute__((ext_vector_type(16)));
typedef _Float16 v8h  __attribute__((ext_vector_type(8)));
typedef float    v8f  __attribute__((ext_vector_type(8)));

#define B_ROWS 16384
#define NUMF   13
#define CFEAT  26
#define VCLS   100000
#define DEMB   16
#define IN_F   429            // 13 + 26*16
#define KPAD   448            // pad to 14 * 32
#define HID    1024
#define ROWS   32             // batch rows per workgroup
#define BLOCK  512            // 16 waves of 32
#define NT     8              // 16x16 N-tiles per wave (128 cols)
#define KT     (KPAD / 32)    // 14 K-steps
#define NTILES (HID / 16)     // 64 N-tiles per K-slice
#define BPART  (NTILES * 512) // halves per {hi|lo} part of one slice buffer (64 KB)
#define LODELTA ((uint64_t)HID * KPAD * 2) // byte distance w1hi -> w1lo in d_ws

// --- Prep: W1 [K=429][N=1024] fp32 -> transposed f16 hi/lo [N=1024][K=448] ---
__global__ __launch_bounds__(256) void prep_w1(const float* __restrict__ W1,
                                               _Float16* __restrict__ w1hi,
                                               _Float16* __restrict__ w1lo) {
  int i = blockIdx.x * 256 + threadIdx.x;
  if (i >= HID * KPAD) return;
  int n = i / KPAD;
  int k = i - n * KPAD;
  float v = (k < IN_F) ? W1[k * HID + n] : 0.0f;
  _Float16 hi = (_Float16)v;
  w1hi[i] = hi;
  w1lo[i] = (_Float16)(v - (float)hi);
}

// Async DMA: 16 bytes global -> LDS, tracked by ASYNCcnt (CDNA5 async path)
__device__ __forceinline__ void async_b128(uint32_t ldsoff, uint64_t gaddr) {
  asm volatile("global_load_async_to_lds_b128 %0, %1, off"
               :: "v"(ldsoff), "v"(gaddr) : "memory");
}
__device__ __forceinline__ void wait_async0() {
  asm volatile("s_wait_asynccnt 0x0" ::: "memory");
}

// A-fragment per ISA 16-bit A layout: lane half h reads halves {k..k+7, k+16..k+23}
__device__ __forceinline__ v16h ld_a16(const _Float16* p) {
  v8h a = *(const v8h*)p;
  v8h b = *(const v8h*)(p + 16);
  return __builtin_shufflevector(a, b, 0,1,2,3,4,5,6,7,8,9,10,11,12,13,14,15);
}
// B-fragment from fragment-ordered LDS tile (p already + lane*8 halves):
// two coalesced 16B chunks at +0 and +512B.
__device__ __forceinline__ v16h ld_b16(const _Float16* p) {
  v8h a = *(const v8h*)p;
  v8h b = *(const v8h*)(p + 256);
  return __builtin_shufflevector(a, b, 0,1,2,3,4,5,6,7,8,9,10,11,12,13,14,15);
}

__global__ __launch_bounds__(BLOCK) void ebm_fused(
    const float* __restrict__ x_num, const int* __restrict__ x_cat,
    const float* __restrict__ emb_w, const float* __restrict__ emb_b,
    const _Float16* __restrict__ w1hi, const _Float16* __restrict__ w1lo,
    const float* __restrict__ b1, const float* __restrict__ W2,
    const float* __restrict__ b2, float* __restrict__ out) {
  __shared__ _Float16 sAhi[ROWS * KPAD];       // 28672 B
  __shared__ _Float16 sAlo[ROWS * KPAD];       // 28672 B
  __shared__ _Float16 sB[2][2 * BPART];        // 2 x 128 KB double-buffered K-slice
  __shared__ float    sRed[ROWS * NT];         // 1 KB

  const int tid = threadIdx.x;
  const int wg  = blockIdx.x;

  // ---- Per-thread staging invariants for the async B slice copy.
  // One K-slice part = 64 tiles x (32 lanes x 2 halves) 16B chunks = 4096
  // chunks; 512 threads -> 8 rounds each for hi and lo.
  uint64_t ghi[8];          // bumped by +64B per slice
  uint32_t ldst0[8], ldst1[8];
#pragma unroll
  for (int rnd = 0; rnd < 8; ++rnd) {
    int c = rnd * BLOCK + tid;        // [0, 4096)
    int t = c >> 6;                   // tile 0..63
    int r = c & 63;
    int l = r & 31;                   // lane slot within tile fragment
    int h = r >> 5;                   // which 16B half of the lane's 32B
    int n = t * 16 + (l & 15);
    int koff = (l >> 4) * 8 + h * 16; // halves within the 32-K slice
    ghi[rnd] = (uint64_t)(uintptr_t)(w1hi + n * KPAD + koff);
    uint32_t off = (uint32_t)(t * 1024 + h * 512 + l * 16); // bytes in part
    ldst0[rnd] = (uint32_t)(uintptr_t)&sB[0][0] + off;
    ldst1[rnd] = (uint32_t)(uintptr_t)&sB[1][0] + off;
  }

  // ---- Kick off async DMA of K-slice 0 into buffer 0 (overlaps the gather).
#pragma unroll
  for (int rnd = 0; rnd < 8; ++rnd) {
    async_b128(ldst0[rnd], ghi[rnd]);                       // hi part
    async_b128(ldst0[rnd] + 65536u, ghi[rnd] + LODELTA);    // lo part
    ghi[rnd] += 64u;                                        // advance to slice 1
  }

  // ---- Phase 1: gather [x_num | per-cat embedding rows] -> f16 hi/lo in LDS
  for (int e = tid; e < ROWS * KPAD; e += BLOCK) {
    int r = e / KPAD;
    int k = e - r * KPAD;
    int row = wg * ROWS + r;
    float v = 0.0f;
    if (k < NUMF) {
      v = x_num[row * NUMF + k];
    } else if (k < IN_F) {
      int j = k - NUMF;
      int c = j >> 4;
      int d = j & 15;
      int idx = x_cat[row * CFEAT + c];
      v = emb_w[(c * VCLS + idx) * DEMB + d] + emb_b[c * DEMB + d];
    }
    _Float16 hi = (_Float16)v;
    sAhi[e] = hi;
    sAlo[e] = (_Float16)(v - (float)hi);
  }
  wait_async0();     // slice 0 resident
  __syncthreads();   // + gather visible to all waves

  // ---- Phase 2: split-f16 WMMA GEMM with async double-buffered B slices
  const int wave = tid >> 5;
  const int lane = tid & 31;
  const int lh   = lane >> 4;
  const int ll   = lane & 15;
  const int mt   = wave >> 3;   // 0..1 : 16-row M tile
  const int nq   = wave & 7;    // 0..7 : 128-col N chunk

  v8f acc[NT] = {};

  const _Float16* aBaseHi = sAhi + (mt * 16 + ll) * KPAD;
  const _Float16* aBaseLo = sAlo + (mt * 16 + ll) * KPAD;

  // Per-wave fragment base pointers: all inner-loop DS accesses become
  // base + small immediate (nt*1024 / nt*1024+512 bytes), so no VALU address
  // math lands between WMMAs (avoids WMMA->VALU hazard NOPs).
  const _Float16* bh0 = sB[0] + nq * NT * 512 + lane * 8;
  const _Float16* bh1 = sB[1] + nq * NT * 512 + lane * 8;
  const _Float16* bl0 = bh0 + BPART;
  const _Float16* bl1 = bh1 + BPART;

  for (int kt = 0; kt < KT; ++kt) {
    // Issue async DMA for slice kt+1 into the other buffer (overlap compute).
    if (kt + 1 < KT) {
      const bool odd = ((kt + 1) & 1) != 0;
#pragma unroll
      for (int rnd = 0; rnd < 8; ++rnd) {
        uint32_t d = odd ? ldst1[rnd] : ldst0[rnd];
        async_b128(d, ghi[rnd]);
        async_b128(d + 65536u, ghi[rnd] + LODELTA);
        ghi[rnd] += 64u;
      }
    }

    const _Float16* bh = (kt & 1) ? bh1 : bh0;
    const _Float16* bl = (kt & 1) ? bl1 : bl0;
    const int ka = kt * 32 + lh * 8;   // A halves {ka..ka+7, ka+16..ka+23}
    v16h ahi = ld_a16(aBaseHi + ka);
    v16h alo = ld_a16(aBaseLo + ka);
#pragma unroll
    for (int nt = 0; nt < NT; ++nt) {
      v16h bhi = ld_b16(bh + nt * 512);
      v16h blo = ld_b16(bl + nt * 512);
      acc[nt] = __builtin_amdgcn_wmma_f32_16x16x32_f16(false, ahi, false, bhi,
                                                       (short)0, acc[nt], false, false);
      acc[nt] = __builtin_amdgcn_wmma_f32_16x16x32_f16(false, alo, false, bhi,
                                                       (short)0, acc[nt], false, false);
      acc[nt] = __builtin_amdgcn_wmma_f32_16x16x32_f16(false, ahi, false, blo,
                                                       (short)0, acc[nt], false, false);
    }

    if (kt + 1 < KT) {
      wait_async0();    // own copies of slice kt+1 done
      __syncthreads();  // all waves' copies done & all done reading buf kt
    }
  }

  // ---- Phase 3: bias + ReLU + dot(W2) in fp32, deterministic reduction
  float psum[8];
#pragma unroll
  for (int i = 0; i < 8; ++i) psum[i] = 0.0f;
#pragma unroll
  for (int nt = 0; nt < NT; ++nt) {
    const int n = nq * 128 + nt * 16 + ll;
    const float bb  = b1[n];
    const float w2v = W2[n];
#pragma unroll
    for (int i = 0; i < 8; ++i) {
      float h = acc[nt][i] + bb;
      h = h > 0.0f ? h : 0.0f;
      psum[i] += h * w2v;
    }
  }
#pragma unroll
  for (int mask = 1; mask < 16; mask <<= 1) {
#pragma unroll
    for (int i = 0; i < 8; ++i) psum[i] += __shfl_xor(psum[i], mask, 16);
  }
  if (ll == 0) {
    const int mbase = mt * 16 + lh * 8;
#pragma unroll
    for (int i = 0; i < 8; ++i) sRed[(mbase + i) * NT + nq] = psum[i];
  }
  __syncthreads();
  if (tid < ROWS) {
    float s = b2[0];
#pragma unroll
    for (int q = 0; q < NT; ++q) s += sRed[tid * NT + q];
    out[wg * ROWS + tid] = s;
  }
}

extern "C" void kernel_launch(void* const* d_in, const int* in_sizes, int n_in,
                              void* d_out, int out_size, void* d_ws, size_t ws_size,
                              hipStream_t stream) {
  const float* x_num = (const float*)d_in[0];
  const int*   x_cat = (const int*)d_in[1];
  const float* emb_w = (const float*)d_in[2];
  const float* emb_b = (const float*)d_in[3];
  const float* W1    = (const float*)d_in[4];
  const float* b1    = (const float*)d_in[5];
  const float* W2    = (const float*)d_in[6];
  const float* b2    = (const float*)d_in[7];
  float* out = (float*)d_out;

  _Float16* w1hi = (_Float16*)d_ws;    // 1024*448 halves
  _Float16* w1lo = w1hi + HID * KPAD;  // 1024*448 halves (contiguous, 1.83 MB)

  const int prepN = HID * KPAD;
  prep_w1<<<(prepN + 255) / 256, 256, 0, stream>>>(W1, w1hi, w1lo);
  ebm_fused<<<B_ROWS / ROWS, BLOCK, 0, stream>>>(x_num, x_cat, emb_w, emb_b,
                                                 w1hi, w1lo, b1, W2, b2, out);
}
